// Perceiver_22162031247366
// MI455X (gfx1250) — matrix operand order repaired and checked by reading, not compiled, not run on hardware
//
#include <hip/hip_runtime.h>
#include <hip/hip_bf16.h>

// ---------------------------------------------------------------------------
// Perceiver cross-attention, MI455X (gfx1250, wave32, WMMA bf16).
//
// Algebraic restructuring:
//   scores[n,h,s] = xn[n,s,:] . qkw[h,:]          (q folded into Wk, K-proj gone)
//   o[n,h,:]      = (att[n,h,:] @ xn[n,:,:]) @ Wv_h^T + bv   (att commutes w/ Wv)
// leaving ~4.3 GFLOP of bf16 WMMA against a 268 MB streaming read of x:
// the kernel is HBM-bound (~12 us floor at 23.3 TB/s).
// ---------------------------------------------------------------------------

typedef __attribute__((ext_vector_type(16))) __bf16 v16bf;
typedef __attribute__((ext_vector_type(8)))  float  v8f;

#define EE   256
#define SS   32
#define NHH  8
#define PIX  8            // pixels (consecutive w) per workgroup
#define LDA  264          // padded row pitch (bf16) for xn rows -> conflict-free A frags

// LDS carve (bytes)
#define OFF_A      0                          // xn  bf16 [256 rows][LDA]
#define OFF_SC     (OFF_A  + 256*LDA*2)       // scores f32 [256][8]
#define OFF_ATT    (OFF_SC + 256*8*4)         // att bf16 [8 pix][16 heads(pad)][32 s]
#define OFF_Y      (OFF_ATT+ 8*16*32*2)       // y   bf16 [8 pix][8 heads][LDA]
#define OFF_O      (OFF_Y  + 8*8*LDA*2)       // o   bf16 [8 pix][LDA]
#define OFF_LNW    (OFF_O  + 8*LDA*2)
#define OFF_LNB    (OFF_LNW+ 256*4)
#define SMEM_BYTES (OFF_LNB+ 256*4)

// workspace carve (bytes)
#define WS_QKW   0                 // bf16 [16][256]  (rows 8..15 zero)
#define WS_WV    8192              // bf16 [256][256]
#define WS_WOUT  (8192 + 131072)   // bf16 [256][256]

__device__ __forceinline__ v8f wmma_bf16(v16bf a, v16bf b, v8f c) {
  return __builtin_amdgcn_wmma_f32_16x16x32_bf16(false, a, false, b, (short)0, c,
                                                 false, false);
}

// A fragment (16x32 bf16): lane holds row m = (lane&15); caller passes
// row pointer already offset by  m*pitch + kBase + 8*(lane>=16).
// comps 0..7 -> +0..7 ; comps 8..15 -> +16..23   (ISA 7.12.2)
__device__ __forceinline__ v16bf load_a_frag(const __bf16* row) {
  v16bf a;
#pragma unroll
  for (int c = 0; c < 8; ++c) a[c] = row[c];
#pragma unroll
  for (int c = 0; c < 8; ++c) a[c + 8] = row[c + 16];
  return a;
}

// B fragment (32x16 bf16), storage N-major (row n contiguous in K): caller
// passes pointer at  n*pitch + kBase + 8*(lane>=16); comps 0..7 -> +0..7 ; comps 8..15 -> +16..23.
__device__ __forceinline__ v16bf load_b_frag(const __bf16* row) {
  v16bf b;
#pragma unroll
  for (int c = 0; c < 8; ++c) b[c] = row[c];
#pragma unroll
  for (int c = 0; c < 8; ++c) b[c + 8] = row[c + 16];
  return b;
}

// ---------------------------------------------------------------------------
// prep1: q = query @ Wq^T + bq ;  qkw[h,:] = scale * sum_d q[h*32+d] * Wk[h*32+d,:]
// ---------------------------------------------------------------------------
__global__ __launch_bounds__(256) void prep1_kernel(const float* __restrict__ query,
                                                    const float* __restrict__ ipw,
                                                    const float* __restrict__ ipb,
                                                    __bf16* __restrict__ qkw) {
  __shared__ float sQ[EE];
  const int t = threadIdx.x;
  float acc = ipb[t];
  const float* row = ipw + t * EE;  // Wq row t
  for (int e = 0; e < EE; ++e) acc += query[e] * row[e];
  sQ[t] = acc;
  __syncthreads();
  const float scale = 0.17677669529663687f;  // 1/sqrt(32)
#pragma unroll
  for (int i = 0; i < 8; ++i) {
    int idx = t + i * 256;           // 0..2047
    int h = idx >> 8, e = idx & 255;
    float a = 0.f;
    const float* wk = ipw + (EE + h * 32) * EE + e;  // Wk rows
    for (int d = 0; d < 32; ++d) a += sQ[h * 32 + d] * wk[d * EE];
    qkw[h * EE + e] = (__bf16)(a * scale);
  }
#pragma unroll
  for (int i = 0; i < 8; ++i) qkw[2048 + i * 256 + t] = (__bf16)0.f;  // pad heads
}

// ---------------------------------------------------------------------------
// prep2: bf16 copies of Wv (in_proj_w rows 512..767) and out_w
// ---------------------------------------------------------------------------
__global__ __launch_bounds__(256) void prep2_kernel(const float* __restrict__ ipw,
                                                    const float* __restrict__ outw,
                                                    __bf16* __restrict__ wvb,
                                                    __bf16* __restrict__ woutb) {
  int gid = blockIdx.x * 256 + threadIdx.x;  // 32768 threads
#pragma unroll
  for (int i = 0; i < 4; ++i) {
    int idx = gid + i * 32768;  // 0..131071
    if (idx < 65536) wvb[idx] = (__bf16)ipw[2 * 65536 + idx];
    else             woutb[idx - 65536] = (__bf16)outw[idx - 65536];
  }
}

// ---------------------------------------------------------------------------
// main fused kernel: 1 WG = 8 pixels (consecutive w), 256 threads = 8 waves
// ---------------------------------------------------------------------------
__global__ __launch_bounds__(256, 1) void perceiver_kernel(
    const float* __restrict__ x, const float* __restrict__ ln_w,
    const float* __restrict__ ln_b, const float* __restrict__ ipb,
    const float* __restrict__ outb, const __bf16* __restrict__ qkw,
    const __bf16* __restrict__ wvb, const __bf16* __restrict__ woutb,
    float* __restrict__ out) {
  extern __shared__ char smem[];
  __bf16* sA   = (__bf16*)(smem + OFF_A);
  float*  sSc  = (float*) (smem + OFF_SC);
  __bf16* sAtt = (__bf16*)(smem + OFF_ATT);
  __bf16* sY   = (__bf16*)(smem + OFF_Y);
  __bf16* sO   = (__bf16*)(smem + OFF_O);
  float*  sLnW = (float*) (smem + OFF_LNW);
  float*  sLnB = (float*) (smem + OFF_LNB);

  const int tid  = threadIdx.x;
  const int lane = tid & 31;
  const int wv   = tid >> 5;           // wave id 0..7
  const int upper = lane >> 4;         // 0 / 1

  const int bid = blockIdx.x;          // 1024 blocks
  const int b   = bid >> 9;
  const int rem = bid & 511;
  const int hh  = rem >> 3;
  const int w0  = (rem & 7) * PIX;

  // ---- phase 0: ln params, zero att pad, stream x + LN stats -> bf16 LDS ----
  sLnW[tid] = ln_w[tid];
  sLnB[tid] = ln_b[tid];
  for (int i = tid; i < 8 * 16 * 32; i += 256) sAtt[i] = (__bf16)0.f;

  const int p = tid & 7;               // pixel within tile
  const int s = tid >> 3;              // sequence position
  const int r = p * SS + s;            // xn row
  const float* xb =
      x + (((size_t)b * EE * SS + s) * 64 + hh) * 64 + (w0 + p);
  __bf16* arow = sA + r * LDA;
  float sum = 0.f, sq = 0.f;
  for (int e = 0; e < EE; ++e) {
    float v = xb[(size_t)e * (SS * 64 * 64)];
    sum += v;
    sq += v * v;
    arow[e] = (__bf16)v;
  }
  const float mu   = sum * (1.f / EE);
  const float var  = sq * (1.f / EE) - mu * mu;
  const float rsig = rsqrtf(var + 1e-5f);
  __syncthreads();
  for (int e = 0; e < EE; ++e) {
    float v = (float)arow[e];
    arow[e] = (__bf16)(((v - mu) * rsig) * sLnW[e] + sLnB[e]);
  }
  __syncthreads();

  // ---- phase 1: scores = xn (256x256) @ qkw^T (256x8, padded 16) ----
  for (int mi = 0; mi < 2; ++mi) {
    const int mt = wv * 2 + mi;        // M-tile 0..15
    v8f acc = {0.f, 0.f, 0.f, 0.f, 0.f, 0.f, 0.f, 0.f};
#pragma unroll
    for (int ks = 0; ks < 8; ++ks) {
      v16bf a = load_a_frag(sA + (mt * 16 + (lane & 15)) * LDA + ks * 32 + upper * 8);
      v16bf bq = load_b_frag(qkw + (lane & 15) * EE + ks * 32 + upper * 8);
      acc = wmma_bf16(a, bq, acc);
    }
    const int n = lane & 15;
    const int mrow = mt * 16 + upper * 8;
    if (n < NHH) {
#pragma unroll
      for (int rr = 0; rr < 8; ++rr) sSc[(mrow + rr) * 8 + n] = acc[rr];
    }
  }
  __syncthreads();

  // ---- phase 2: softmax over s (per pixel,head) -> bf16 att ----
  if (tid < 64) {
    const int pp = tid >> 3, h = tid & 7;
    float mx = -1e30f;
#pragma unroll
    for (int s2 = 0; s2 < SS; ++s2)
      mx = fmaxf(mx, sSc[(pp * SS + s2) * 8 + h]);
    float ex[SS], sm = 0.f;
#pragma unroll
    for (int s2 = 0; s2 < SS; ++s2) {
      ex[s2] = __expf(sSc[(pp * SS + s2) * 8 + h] - mx);
      sm += ex[s2];
    }
    const float inv = 1.f / sm;
#pragma unroll
    for (int s2 = 0; s2 < SS; ++s2)
      sAtt[(pp * 16 + h) * SS + s2] = (__bf16)(ex[s2] * inv);
  }
  __syncthreads();

  // ---- phase 3: y[p] = att[p] (8x32, padded 16) @ xn[p] (32x256) ----
  {
    const int pp = wv;                 // pixel per wave
    v16bf a;                           // att fragment, K = s
    {
      const __bf16* ar = sAtt + (pp * 16 + (lane & 15)) * SS + upper * 8;
#pragma unroll
      for (int c = 0; c < 8; ++c) a[c] = ar[c];
#pragma unroll
      for (int c = 0; c < 8; ++c) a[c + 8] = ar[c + 16];
    }
    for (int et = 0; et < 16; ++et) {
      v16bf bb;                        // xn as B: B[k=s][n=e], s-major in rows
      const int e = et * 16 + (lane & 15);
#pragma unroll
      for (int c = 0; c < 16; ++c)
        bb[c] = sA[(pp * SS + upper * 8 + c + (c & 8)) * LDA + e];
      v8f acc = {0.f, 0.f, 0.f, 0.f, 0.f, 0.f, 0.f, 0.f};
      acc = wmma_bf16(a, bb, acc);
      if (lane < 16) {
#pragma unroll
        for (int rr = 0; rr < 8; ++rr)   // rr = head
          sY[(pp * 8 + rr) * LDA + et * 16 + lane] = (__bf16)acc[rr];
      }
    }
  }
  __syncthreads();

  // ---- phase 4: o[:, h*32..] = Y_h (8x256, pad 16) @ Wv_h^T (256x32) + bv ----
  {
    const int h = wv;                  // head per wave
    for (int nt = 0; nt < 2; ++nt) {
      v8f acc = {0.f, 0.f, 0.f, 0.f, 0.f, 0.f, 0.f, 0.f};
#pragma unroll
      for (int ks = 0; ks < 8; ++ks) {
        const int pp = (lane & 15) & 7;  // M rows 8..15 duplicate (unused)
        v16bf a = load_a_frag(sY + (pp * 8 + h) * LDA + ks * 32 + upper * 8);
        const int j = h * 32 + nt * 16 + (lane & 15);
        v16bf bb = load_b_frag(wvb + j * EE + ks * 32 + upper * 8);
        acc = wmma_bf16(a, bb, acc);
      }
      const int j = h * 32 + nt * 16 + (lane & 15);
      const float bvj = ipb[2 * EE + j];
      if (lane < 16) {
#pragma unroll
        for (int rr = 0; rr < 8; ++rr)   // rr = pixel
          sO[rr * LDA + j] = (__bf16)(acc[rr] + bvj);
      }
    }
  }
  __syncthreads();

  // ---- phase 5: out = o (8x256, pad 16) @ out_w^T (256x256) + out_b ----
  for (int ni = 0; ni < 2; ++ni) {
    const int nt = wv * 2 + ni;        // N-tile 0..15
    v8f acc = {0.f, 0.f, 0.f, 0.f, 0.f, 0.f, 0.f, 0.f};
#pragma unroll
    for (int ks = 0; ks < 8; ++ks) {
      const int pp = (lane & 15) & 7;
      v16bf a = load_a_frag(sO + pp * LDA + ks * 32 + upper * 8);
      const int j2 = nt * 16 + (lane & 15);
      v16bf bb = load_b_frag(woutb + j2 * EE + ks * 32 + upper * 8);
      acc = wmma_bf16(a, bb, acc);
    }
    const int j2 = nt * 16 + (lane & 15);
    const float ob = outb[j2];
    if (lane < 16) {
#pragma unroll
      for (int rr = 0; rr < 8; ++rr) {   // rr = pixel
        out[((size_t)(b * EE + j2) * 64 + hh) * 64 + (w0 + rr)] = acc[rr] + ob;
      }
    }
  }
}

// ---------------------------------------------------------------------------
extern "C" void kernel_launch(void* const* d_in, const int* in_sizes, int n_in,
                              void* d_out, int out_size, void* d_ws, size_t ws_size,
                              hipStream_t stream) {
  const float* x     = (const float*)d_in[0];
  const float* query = (const float*)d_in[1];
  const float* ln_w  = (const float*)d_in[2];
  const float* ln_b  = (const float*)d_in[3];
  const float* ipw   = (const float*)d_in[4];
  const float* ipb   = (const float*)d_in[5];
  const float* outw  = (const float*)d_in[6];
  const float* outb  = (const float*)d_in[7];
  float* out = (float*)d_out;

  __bf16* qkw   = (__bf16*)((char*)d_ws + WS_QKW);
  __bf16* wvb   = (__bf16*)((char*)d_ws + WS_WV);
  __bf16* woutb = (__bf16*)((char*)d_ws + WS_WOUT);

  prep1_kernel<<<1, 256, 0, stream>>>(query, ipw, ipb, qkw);
  prep2_kernel<<<128, 256, 0, stream>>>(ipw, outw, wvb, woutb);

  const int nBlocks = 2 * 64 * (64 / PIX);  // 1024
  perceiver_kernel<<<nBlocks, 256, SMEM_BYTES, stream>>>(
      x, ln_w, ln_b, ipb, outb, qkw, wvb, woutb, out);
}
